// MQTokenizer_56229711839526
// MI455X (gfx1250) — matrix operand order repaired and checked by reading, not compiled, not run on hardware
//
#include <hip/hip_runtime.h>
#include <cmath>

typedef _Float16 h16;
typedef __attribute__((ext_vector_type(16))) _Float16 v16h;
typedef __attribute__((ext_vector_type(8)))  float    v8f;
typedef __attribute__((ext_vector_type(4)))  unsigned u32x4;
typedef __attribute__((ext_vector_type(8)))  unsigned u32x8;

#define DEVI __device__ __forceinline__

DEVI v8f wmma_f16(v16h a, v16h b, v8f c) {
  // D = A(16x32 f16) * B(32x16 f16) + C(16x16 f32)
  return __builtin_amdgcn_wmma_f32_16x16x32_f16(false, a, false, b, (short)0, c,
                                                false, false);
}

DEVI v8f vzero8() {
  v8f z;
#pragma unroll
  for (int t = 0; t < 8; ++t) z[t] = 0.0f;
  return z;
}

// A fragment (16x32, f16) from row-major LDS tile (stride in halves).
// Lane layout per CDNA5 ISA: lanes 0-15 -> K{0..7,16..23}, lanes 16-31 -> K{8..15,24..31}.
DEVI v16h lds_load_a(const h16* X, int stride, int m0, int k0) {
  const int lane = threadIdx.x & 31;
  const h16* p = X + (m0 + (lane & 15)) * stride + k0 + ((lane >> 4) << 3);
  v16h a;
#pragma unroll
  for (int t = 0; t < 8; ++t) { a[t] = p[t]; a[8 + t] = p[16 + t]; }
  return a;
}

// B fragment (32x16, f16) from transposed (N x K) LDS tile.
// Lane n holds K values: lanes 0-15 -> K{0..7,16..23}, lanes 16-31 -> K{8..15,24..31}.
DEVI v16h lds_load_b(const h16* Wt, int stride, int n0, int k0) {
  const int lane = threadIdx.x & 31;
  const h16* p = Wt + (n0 + (lane & 15)) * stride + k0 + ((lane >> 4) << 3);
  v16h b;
#pragma unroll
  for (int t = 0; t < 8; ++t) { b[t] = p[t]; b[8 + t] = p[16 + t]; }
  return b;
}

// ---------------------------------------------------------------------------
// Kernel Z: zero avg_q accumulator and the three loss scalars.
__global__ __launch_bounds__(256) void mq_zero(float* __restrict__ avgq,
                                               float* __restrict__ losses) {
  size_t i = (size_t)blockIdx.x * 256 + threadIdx.x;
  if (i < (size_t)65536 * 64) avgq[i] = 0.0f;
  if (i < 3) losses[i] = 0.0f;
}

// ---------------------------------------------------------------------------
// Kernel 0: L2-normalize codebooks -> f32 (for q / losses) + f16 (WMMA B operand).
__global__ __launch_bounds__(64) void mq_norm_cb(const float* __restrict__ cb,
                                                 float* __restrict__ outf,
                                                 h16* __restrict__ outh) {
  __shared__ float red[64];
  const int row = blockIdx.x, t = threadIdx.x;
  float v = cb[(size_t)row * 64 + t];
  red[t] = v * v;
  __syncthreads();
  for (int s = 32; s > 0; s >>= 1) {
    if (t < s) red[t] += red[t + s];
    __syncthreads();
  }
  float scale = 1.0f / fmaxf(sqrtf(red[0]), 1e-12f);
  float nv = v * scale;
  outf[(size_t)row * 64 + t] = nv;
  outh[(size_t)row * 64 + t] = (h16)nv;
}

// ---------------------------------------------------------------------------
// Kernel A: fused encoder for one sub-quantizer k over a 64-row tile.
// grid = (B/64, K), block = 256 (8 waves).
__global__ __launch_bounds__(256) void mq_encoder(
    const float* __restrict__ x, const float* __restrict__ W1,
    const float* __restrict__ b1, const float* __restrict__ W2,
    const float* __restrict__ b2, const float* __restrict__ W3,
    const float* __restrict__ b3, const float* __restrict__ gamma,
    const float* __restrict__ beta, const h16* __restrict__ cbn_h,
    const float* __restrict__ cbn_f, float* __restrict__ avgq,
    float* __restrict__ tokens, float* __restrict__ losses) {
  __shared__ __align__(16) unsigned char pool[52480];

  const int tid = threadIdx.x;
  const int lane = tid & 31;
  const int wid = tid >> 5;
  const int r = wid >> 1;  // row-block 0..3 (16 rows each)
  const int h = wid & 1;   // column half
  const int kq = blockIdx.y;
  const int m_base = blockIdx.x * 64;

  const float* W1k = W1 + (size_t)kq * 512 * 256;
  const float* b1k = b1 + (size_t)kq * 256;
  const float* W2k = W2 + (size_t)kq * 256 * 128;
  const float* b2k = b2 + (size_t)kq * 128;
  const float* W3k = W3 + (size_t)kq * 128 * 64;
  const float* b3k = b3 + (size_t)kq * 64;

  // ------------------ Layer 1: h1 = relu(x @ W1 + b1), 64x256 -------------
  {
    h16* Xs  = (h16*)pool;                 // 64 x 72 halves
    h16* Ws1 = (h16*)(pool + 9216);        // 256 x 72 halves
    v8f acc1[8];
#pragma unroll
    for (int nb = 0; nb < 8; ++nb) acc1[nb] = vzero8();

    for (int kc = 0; kc < 512; kc += 64) {
      // pull next weight chunk ahead (global_prefetch_b8)
      if (kc + 64 < 512) {
        const char* nxt = (const char*)(W1k + (size_t)(kc + 64) * 256);
        __builtin_prefetch(nxt + (size_t)tid * 256, 0, 1);
      }
      for (int i = tid; i < 64 * 64; i += 256) {
        int rr = i >> 6, cc = i & 63;
        Xs[rr * 72 + cc] = (h16)x[(size_t)(m_base + rr) * 512 + kc + cc];
      }
      for (int i = tid; i < 64 * 256; i += 256) {
        int kk = i >> 8, n = i & 255;
        Ws1[n * 72 + kk] = (h16)W1k[(size_t)(kc + kk) * 256 + n];
      }
      __syncthreads();
      v16h a0 = lds_load_a(Xs, 72, 16 * r, 0);
      v16h a1 = lds_load_a(Xs, 72, 16 * r, 32);
#pragma unroll
      for (int nb = 0; nb < 8; ++nb) {
        int n0 = h * 128 + nb * 16;
        v16h bb0 = lds_load_b(Ws1, 72, n0, 0);
        v16h bb1 = lds_load_b(Ws1, 72, n0, 32);
        acc1[nb] = wmma_f16(a0, bb0, acc1[nb]);
        acc1[nb] = wmma_f16(a1, bb1, acc1[nb]);
      }
      __syncthreads();
    }
    h16* H1 = (h16*)pool;  // 64 x 264 halves
#pragma unroll
    for (int nb = 0; nb < 8; ++nb) {
      int n = h * 128 + nb * 16 + (lane & 15);
      float bias = b1k[n];
#pragma unroll
      for (int rr = 0; rr < 8; ++rr) {
        int m = 16 * r + ((lane >> 4) << 3) + rr;
        H1[m * 264 + n] = (h16)fmaxf(acc1[nb][rr] + bias, 0.0f);
      }
    }
    __syncthreads();
  }

  // ------------------ Layer 2: h2 = relu(h1 @ W2 + b2), 64x128 ------------
  {
    const h16* H1 = (const h16*)pool;      // 64 x 264
    h16* Ws2 = (h16*)(pool + 33792);       // 128 x 72
    v8f acc2[4];
#pragma unroll
    for (int nb = 0; nb < 4; ++nb) acc2[nb] = vzero8();

    for (int kc = 0; kc < 256; kc += 64) {
      for (int i = tid; i < 64 * 128; i += 256) {
        int kk = i >> 7, n = i & 127;
        Ws2[n * 72 + kk] = (h16)W2k[(size_t)(kc + kk) * 128 + n];
      }
      __syncthreads();
      v16h a0 = lds_load_a(H1, 264, 16 * r, kc);
      v16h a1 = lds_load_a(H1, 264, 16 * r, kc + 32);
#pragma unroll
      for (int nb = 0; nb < 4; ++nb) {
        int n0 = h * 64 + nb * 16;
        v16h bb0 = lds_load_b(Ws2, 72, n0, 0);
        v16h bb1 = lds_load_b(Ws2, 72, n0, 32);
        acc2[nb] = wmma_f16(a0, bb0, acc2[nb]);
        acc2[nb] = wmma_f16(a1, bb1, acc2[nb]);
      }
      __syncthreads();
    }
    h16* H2 = (h16*)pool;  // 64 x 136
#pragma unroll
    for (int nb = 0; nb < 4; ++nb) {
      int n = h * 64 + nb * 16 + (lane & 15);
      float bias = b2k[n];
#pragma unroll
      for (int rr = 0; rr < 8; ++rr) {
        int m = 16 * r + ((lane >> 4) << 3) + rr;
        H2[m * 136 + n] = (h16)fmaxf(acc2[nb][rr] + bias, 0.0f);
      }
    }
    __syncthreads();
  }

  float* encb = (float*)(pool + 34816);  // 64 x 68 floats (post-LN enc)

  // ------------------ Layer 3: enc = h2 @ W3 + b3, 64x64 ------------------
  {
    const h16* H2 = (const h16*)pool;      // 64 x 136
    h16* Ws3 = (h16*)(pool + 17408);       // 64 x 136
    for (int i = tid; i < 128 * 64; i += 256) {
      int kk = i >> 6, n = i & 63;
      Ws3[n * 136 + kk] = (h16)W3k[(size_t)kk * 64 + n];
    }
    __syncthreads();
    const int t0 = wid * 2;
    const int r3 = t0 >> 2, nbb = t0 & 3;
    v16h a[4];
#pragma unroll
    for (int i = 0; i < 4; ++i) a[i] = lds_load_a(H2, 136, 16 * r3, i * 32);
#pragma unroll
    for (int tt = 0; tt < 2; ++tt) {
      int n0 = (nbb + tt) * 16;
      v8f c = vzero8();
#pragma unroll
      for (int i = 0; i < 4; ++i) {
        v16h bb = lds_load_b(Ws3, 136, n0, i * 32);
        c = wmma_f16(a[i], bb, c);
      }
      int n = n0 + (lane & 15);
      float bias = b3k[n];
#pragma unroll
      for (int rr = 0; rr < 8; ++rr) {
        int m = 16 * r3 + ((lane >> 4) << 3) + rr;
        encb[m * 68 + n] = c[rr] + bias;
      }
    }
    __syncthreads();
  }

  // ------------------ LayerNorm + L2 normalize ----------------------------
  h16* ENn = (h16*)pool;  // 64 x 72 halves (enc_n, f16)
  if (tid < 64) {
    float* row = encb + tid * 68;
    float s = 0.0f;
    for (int j = 0; j < 64; ++j) s += row[j];
    float mu = s * (1.0f / 64.0f);
    float s2 = 0.0f;
    for (int j = 0; j < 64; ++j) { float d = row[j] - mu; s2 += d * d; }
    float rs = rsqrtf(s2 * (1.0f / 64.0f) + 1e-5f);
    float nrm = 0.0f;
    for (int j = 0; j < 64; ++j) {
      float v = (row[j] - mu) * rs * gamma[j] + beta[j];
      row[j] = v;
      nrm += v * v;
    }
    float scale = 1.0f / fmaxf(sqrtf(nrm), 1e-12f);
    h16* en = ENn + tid * 72;
    for (int j = 0; j < 64; ++j) en[j] = (h16)(row[j] * scale);
  }
  __syncthreads();

  // ------------------ Cosine GEMM vs codebook + argmax --------------------
  h16*   CBs   = (h16*)(pool + 9216);    // 64 x 72 (144B row stride)
  float* bestV = (float*)(pool + 18432); // 64 x 32
  int*   bestI = (int*)(pool + 26624);   // 64 x 32
  int*   idxA  = (int*)(pool + 52224);   // 64

  const unsigned cbs_lds = (unsigned)(size_t)(void*)CBs;  // LDS byte offset

  float bv[8];
  int   bi[8];
#pragma unroll
  for (int rr = 0; rr < 8; ++rr) { bv[rr] = -1e30f; bi[rr] = 0; }
  v16h ea0 = lds_load_a(ENn, 72, 16 * r, 0);
  v16h ea1 = lds_load_a(ENn, 72, 16 * r, 32);

  for (int ch = 0; ch < 8; ++ch) {
    // Stage 64 x 128B codebook rows into LDS (row stride 144B) via the
    // Tensor Data Mover: one DMA descriptor, issued by wave 0 only
    // (TDM ignores EXEC). Padding fields add 16B per 128B row.
    if (wid == 0) {
      unsigned long long ga =
          (unsigned long long)(size_t)(cbn_h +
                                       ((size_t)kq * 512 + ch * 64) * 64);
      u32x4 g0;
      g0[0] = 1u;                                   // count=1 (valid user D#)
      g0[1] = cbs_lds;                              // lds_addr
      g0[2] = (unsigned)(ga & 0xFFFFFFFFu);         // global_addr[31:0]
      g0[3] = (unsigned)((ga >> 32) & 0x01FFFFFFu)  // global_addr[56:32]
              | 0x80000000u;                        // type=2 ("image")
      u32x8 g1;
      g1[0] = (3u << 16)      // data_size = 8B units
            | (1u << 20)      // pad_enable
            | (4u << 22)      // pad_interval: 32 DWORDs (=128B)
            | (3u << 25);     // pad_amount: 4 DWORDs (=16B)
      g1[1] = (16u << 16);    // tensor_dim0 = 16 units (128B row)
      g1[2] = (512u << 16);   // tensor_dim1 = 512 rows
      g1[3] = (16u << 16);    // tile_dim0 = 16 units
      g1[4] = 64u;            // tile_dim1 = 64 rows (tile_dim2 = 0)
      g1[5] = 16u;            // tensor_dim0_stride = 16 units
      g1[6] = 0u;
      g1[7] = 0u;
      u32x4 gz;
      gz[0] = 0u; gz[1] = 0u; gz[2] = 0u; gz[3] = 0u;
      asm volatile("tensor_load_to_lds %0, %1, %2, %3"
                   :
                   : "s"(g0), "s"(g1), "s"(gz), "s"(gz)
                   : "memory");
      __builtin_amdgcn_s_wait_tensorcnt(0);
    }
    __syncthreads();
#pragma unroll
    for (int nb = 0; nb < 2; ++nb) {
      int n0 = h * 32 + nb * 16;
      v16h bb0 = lds_load_b(CBs, 72, n0, 0);
      v16h bb1 = lds_load_b(CBs, 72, n0, 32);
      v8f c = vzero8();
      c = wmma_f16(ea0, bb0, c);
      c = wmma_f16(ea1, bb1, c);
      int col = ch * 64 + n0 + (lane & 15);
#pragma unroll
      for (int rr = 0; rr < 8; ++rr) {
        if (c[rr] > bv[rr]) { bv[rr] = c[rr]; bi[rr] = col; }
      }
    }
    __syncthreads();
  }
#pragma unroll
  for (int rr = 0; rr < 8; ++rr) {
    int m = 16 * r + ((lane >> 4) << 3) + rr;
    int slot = h * 16 + (lane & 15);
    bestV[m * 32 + slot] = bv[rr];
    bestI[m * 32 + slot] = bi[rr];
  }
  __syncthreads();
  if (tid < 64) {
    const float* pv = bestV + tid * 32;
    const int*   pi = bestI + tid * 32;
    float best = pv[0];
    int bidx = pi[0];
    for (int s = 1; s < 32; ++s) {
      if (pv[s] > best) { best = pv[s]; bidx = pi[s]; }
    }
    idxA[tid] = bidx;
    tokens[(size_t)(m_base + tid) * 3 + kq] = (float)bidx;
  }
  __syncthreads();

  // ------------------ q gather, avg_q accumulate, losses ------------------
  {
    const int row = tid >> 2, part = tid & 3;
    const int idx = idxA[row];
    const float* qv = cbn_f + ((size_t)kq * 512 + idx) * 64 + part * 16;
    const float* er = encb + row * 68 + part * 16;
    float* av = avgq + (size_t)(m_base + row) * 64 + part * 16;
    float ls = 0.0f;
#pragma unroll
    for (int t = 0; t < 16; ++t) {
      float qn = qv[t];
      float d = er[t] - qn;
      ls += d * d;
      atomicAdd(&av[t], qn * (1.0f / 3.0f));
    }
    float* red = (float*)pool;
    red[tid] = ls;
    __syncthreads();
    if (tid == 0) {
      float s = 0.0f;
      for (int i = 0; i < 256; ++i) s += red[i];
      s *= (1.0f / (65536.0f * 64.0f));
      atomicAdd(&losses[1], s);  // codebook loss
      atomicAdd(&losses[2], s);  // commitment loss (numerically identical fwd)
    }
  }
}

// ---------------------------------------------------------------------------
// Kernel B: decoder avg_q -> 128 -> 256 -> 512, recon + recon_loss.
// grid = B/32, block = 256 (8 waves).
__global__ __launch_bounds__(256) void mq_decoder(
    const float* __restrict__ avgq, const float* __restrict__ D1,
    const float* __restrict__ c1, const float* __restrict__ D2,
    const float* __restrict__ c2, const float* __restrict__ D3,
    const float* __restrict__ c3, const float* __restrict__ x,
    float* __restrict__ recon, float* __restrict__ losses) {
  __shared__ __align__(16) unsigned char pool[57344];
  const int tid = threadIdx.x;
  const int lane = tid & 31;
  const int wid = tid >> 5;
  const int m_base = blockIdx.x * 32;

  h16* Xd  = (h16*)pool;                 // 32 x 72
  h16* D1t = (h16*)(pool + 4608);        // 128 x 72
  h16* H1d = (h16*)pool;                 // 32 x 136 (after GEMM1)
  h16* D2s = (h16*)(pool + 8704);        // 64 x 136
  h16* H2d = (h16*)(pool + 40448);       // 32 x 264
  h16* D3s = (h16*)pool;                 // 64 x 264
  float* red = (float*)(pool + 33792);   // 256 floats

  for (int i = tid; i < 32 * 64; i += 256) {
    int rr = i >> 6, cc = i & 63;
    Xd[rr * 72 + cc] = (h16)avgq[(size_t)(m_base + rr) * 64 + cc];
  }
  for (int i = tid; i < 64 * 128; i += 256) {
    int kk = i >> 7, n = i & 127;
    D1t[n * 72 + kk] = (h16)D1[(size_t)kk * 128 + n];
  }
  __syncthreads();

  const int rd = wid >> 2;  // row-block 0..1
  const int qd = wid & 3;   // column quarter

  // GEMM 1: 32x64 @ 64x128
  {
    v16h a0 = lds_load_a(Xd, 72, 16 * rd, 0);
    v16h a1 = lds_load_a(Xd, 72, 16 * rd, 32);
    v8f acc[2];
#pragma unroll
    for (int nb = 0; nb < 2; ++nb) {
      int n0 = qd * 32 + nb * 16;
      v16h bb0 = lds_load_b(D1t, 72, n0, 0);
      v16h bb1 = lds_load_b(D1t, 72, n0, 32);
      v8f c = vzero8();
      c = wmma_f16(a0, bb0, c);
      c = wmma_f16(a1, bb1, c);
      acc[nb] = c;
    }
    __syncthreads();
#pragma unroll
    for (int nb = 0; nb < 2; ++nb) {
      int n = qd * 32 + nb * 16 + (lane & 15);
      float bias = c1[n];
#pragma unroll
      for (int rr = 0; rr < 8; ++rr) {
        int m = 16 * rd + ((lane >> 4) << 3) + rr;
        H1d[m * 136 + n] = (h16)fmaxf(acc[nb][rr] + bias, 0.0f);
      }
    }
    __syncthreads();
  }

  // GEMM 2: 32x128 @ 128x256, N chunked by 64
  {
    v16h a[4];
#pragma unroll
    for (int i = 0; i < 4; ++i) a[i] = lds_load_a(H1d, 136, 16 * rd, i * 32);
    for (int ch = 0; ch < 4; ++ch) {
      for (int i = tid; i < 128 * 64; i += 256) {
        int kk = i >> 6, n = i & 63;
        D2s[n * 136 + kk] = (h16)D2[(size_t)kk * 256 + ch * 64 + n];
      }
      __syncthreads();
      int n0 = qd * 16;
      v8f c = vzero8();
#pragma unroll
      for (int i = 0; i < 4; ++i) {
        v16h bb = lds_load_b(D2s, 136, n0, i * 32);
        c = wmma_f16(a[i], bb, c);
      }
      int ng = ch * 64 + n0 + (lane & 15);
      float bias = c2[ng];
#pragma unroll
      for (int rr = 0; rr < 8; ++rr) {
        int m = 16 * rd + ((lane >> 4) << 3) + rr;
        H2d[m * 264 + ng] = (h16)fmaxf(c[rr] + bias, 0.0f);
      }
      __syncthreads();
    }
  }

  // GEMM 3: 32x256 @ 256x512, N chunked by 64; epilogue writes recon + loss
  {
    v16h a[8];
#pragma unroll
    for (int i = 0; i < 8; ++i) a[i] = lds_load_a(H2d, 264, 16 * rd, i * 32);
    float lsum = 0.0f;
    for (int ch = 0; ch < 8; ++ch) {
      for (int i = tid; i < 256 * 64; i += 256) {
        int kk = i >> 6, n = i & 63;
        D3s[n * 264 + kk] = (h16)D3[(size_t)kk * 512 + ch * 64 + n];
      }
      __syncthreads();
      int n0 = qd * 16;
      v8f c = vzero8();
#pragma unroll
      for (int i = 0; i < 8; ++i) {
        v16h bb = lds_load_b(D3s, 264, n0, i * 32);
        c = wmma_f16(a[i], bb, c);
      }
      int ng = ch * 64 + n0 + (lane & 15);
      float bias = c3[ng];
#pragma unroll
      for (int rr = 0; rr < 8; ++rr) {
        int m = 16 * rd + ((lane >> 4) << 3) + rr;
        size_t g = (size_t)(m_base + m) * 512 + ng;
        float v = c[rr] + bias;
        recon[g] = v;
        float d = v - x[g];
        lsum += d * d;
      }
      __syncthreads();
    }
    red[tid] = lsum;
    __syncthreads();
    if (tid == 0) {
      float s = 0.0f;
      for (int i = 0; i < 256; ++i) s += red[i];
      atomicAdd(&losses[0], s * (1.0f / (65536.0f * 512.0f)));
    }
  }
}

// ---------------------------------------------------------------------------
extern "C" void kernel_launch(void* const* d_in, const int* in_sizes, int n_in,
                              void* d_out, int out_size, void* d_ws,
                              size_t ws_size, hipStream_t stream) {
  (void)in_sizes; (void)n_in; (void)out_size; (void)ws_size;

  const float* x     = (const float*)d_in[0];
  const float* W1    = (const float*)d_in[1];
  const float* b1    = (const float*)d_in[2];
  const float* W2    = (const float*)d_in[3];
  const float* b2    = (const float*)d_in[4];
  const float* W3    = (const float*)d_in[5];
  const float* b3    = (const float*)d_in[6];
  const float* gamma = (const float*)d_in[7];
  const float* beta  = (const float*)d_in[8];
  const float* cbks  = (const float*)d_in[9];
  const float* D1    = (const float*)d_in[10];
  const float* c1    = (const float*)d_in[11];
  const float* D2    = (const float*)d_in[12];
  const float* c2    = (const float*)d_in[13];
  const float* D3    = (const float*)d_in[14];
  const float* c3    = (const float*)d_in[15];

  float* out    = (float*)d_out;
  float* tokens = out;                         // (B, K) = 65536*3
  float* recon  = out + 196608;                // (B, 512)
  float* losses = out + 33751040;              // [recon, codebook, commitment]

  float* avgq  = (float*)d_ws;                 // B*64 f32 accumulator
  float* cbn_f = avgq + (size_t)65536 * 64;    // K*512*64 f32
  h16*   cbn_h = (h16*)(cbn_f + (size_t)3 * 512 * 64);  // K*512*64 f16

  mq_zero<<<16384, 256, 0, stream>>>(avgq, losses);
  mq_norm_cb<<<1536, 64, 0, stream>>>(cbks, cbn_f, cbn_h);

  dim3 gA(1024, 3);
  mq_encoder<<<gA, 256, 0, stream>>>(x, W1, b1, W2, b2, W3, b3, gamma, beta,
                                     cbn_h, cbn_f, avgq, tokens, losses);
  mq_decoder<<<2048, 256, 0, stream>>>(avgq, D1, c1, D2, c2, D3, c3, x, recon,
                                       losses);
}
